// BigBirdProteinModel_71708773974873
// MI455X (gfx1250) — compile-verified, matrix operand order checked by reading, not run on hardware
//
#include <hip/hip_runtime.h>

// ---------------------------------------------------------------------------
// Types for CDNA5 WMMA / TDM (gfx1250, wave32)
// ---------------------------------------------------------------------------
typedef __attribute__((ext_vector_type(16))) __bf16 v16bf;
typedef __attribute__((ext_vector_type(8)))  __bf16 v8bf;
typedef __attribute__((ext_vector_type(8)))  float  v8f;
typedef __attribute__((ext_vector_type(4)))  unsigned int u32x4;
typedef __attribute__((ext_vector_type(4)))  int i32x4;
typedef __attribute__((ext_vector_type(8)))  int i32x8;

#define WMMA_BF16(a, b, c) \
    __builtin_amdgcn_wmma_f32_16x16x32_bf16(false, (a), false, (b), (short)0, (c), false, false)

// Build a 16-element bf16 fragment from two contiguous 8-element (16B) chunks.
// ISA 7.12.2 (16-bit A/B fragment): elements 0..7 = K{koff..+7},
// elements 8..15 = K{koff+16..+23}, koff = (lane<16)?0:8.
__device__ inline v16bf make_frag(const __bf16* p0, const __bf16* p1) {
    v16bf r;
    v8bf lo = *(const v8bf*)p0;
    v8bf hi = *(const v8bf*)p1;
#pragma unroll
    for (int i = 0; i < 8; ++i) { r[i] = lo[i]; r[i + 8] = hi[i]; }
    return r;
}

__device__ inline float gelu_new_f(float x) {
    const float c = 0.7978845608028654f;
    return 0.5f * x * (1.0f + tanhf(c * (x + 0.044715f * x * x * x)));
}

// ---------------------------------------------------------------------------
// Tensor Data Mover: async 2D tile load (global -> LDS).
// Builds a D# per CDNA5 ISA: group0 {count=1, lds_addr,
// global_addr, type=2}; group1 {data_size=2B, tensor_dim0/1, tile_dim0/1,
// tensor_dim0_stride}. Groups 2/3 zero (2D tensor). Tracked by TENSORcnt.
// ---------------------------------------------------------------------------
__device__ inline void tdm_load_2d(unsigned long long gaddr, unsigned lds_off,
                                   unsigned tileW /*elems*/, unsigned tileH,
                                   unsigned rowStride /*elems*/) {
    u32x4 g0;
    g0[0] = 1u;                                        // count=1 (valid), user mode
    g0[1] = lds_off;                                   // lds_addr (bytes)
    g0[2] = (unsigned)(gaddr & 0xffffffffull);         // global_addr[31:0]
    g0[3] = (unsigned)((gaddr >> 32) & 0x01ffffffull)  // global_addr[56:32]
            | 0x80000000u;                             // type=2 ("image") @bits127:126
    i32x8 g1;
    g1[0] = (int)(1u << 16);                           // data_size=1 -> 2 bytes/elem
    g1[1] = (int)((tileW & 0xffffu) << 16);            // tensor_dim0[15:0] @ bits63:48
    g1[2] = (int)((tileW >> 16) | ((tileH & 0xffffu) << 16)); // dim0 hi | tensor_dim1 lo
    g1[3] = (int)((tileH >> 16) | (tileW << 16));      // tensor_dim1 hi | tile_dim0
    g1[4] = (int)tileH;                                // tile_dim1 (tile_dim2 = 0)
    g1[5] = (int)rowStride;                            // tensor_dim0_stride[31:0]
    g1[6] = 0;
    g1[7] = 0;
    i32x4 gz = {0, 0, 0, 0};
#if defined(__clang_major__) && (__clang_major__ >= 23)
    i32x8 gz8 = {0, 0, 0, 0, 0, 0, 0, 0};
    __builtin_amdgcn_tensor_load_to_lds(g0, g1, gz, gz, gz8, 0);
#else
    __builtin_amdgcn_tensor_load_to_lds(g0, g1, gz, gz, 0);
#endif
}

// ---------------------------------------------------------------------------
// f32 -> bf16 conversion (grid-stride)
// ---------------------------------------------------------------------------
__global__ void cvt_f32_bf16(const float* __restrict__ src, __bf16* __restrict__ dst, long n) {
    long i = (long)blockIdx.x * blockDim.x + threadIdx.x;
    long stride = (long)gridDim.x * blockDim.x;
    for (; i < n; i += stride) dst[i] = (__bf16)src[i];
}

// ---------------------------------------------------------------------------
// f32 [K x N] -> bf16 transposed [N x K] (tiled through LDS, coalesced).
// ---------------------------------------------------------------------------
__global__ __launch_bounds__(256) void cvt_transpose(
    const float* __restrict__ src, __bf16* __restrict__ dst, int Kd, int Nd) {
    __shared__ float tile[32][33];
    int k0 = blockIdx.x * 32, n0 = blockIdx.y * 32;
    int tx = threadIdx.x, ty = threadIdx.y;
#pragma unroll
    for (int i = 0; i < 4; ++i)
        tile[ty + i * 8][tx] = src[(size_t)(k0 + ty + i * 8) * Nd + n0 + tx];
    __syncthreads();
#pragma unroll
    for (int i = 0; i < 4; ++i)
        dst[(size_t)(n0 + ty + i * 8) * Kd + k0 + tx] = (__bf16)tile[tx][ty + i * 8];
}

// ---------------------------------------------------------------------------
// bf16 WMMA GEMM: out = act(A[MxK] @ Bt[NxK]^T + bias) (+ resid)
// Workgroup tile 128x128, 8 waves (2x4), each wave 64x32 (4x2 accumulators).
// Double-buffered LDS, vectorized 16B staging, global prefetch for tile t+2.
// ACT: 0=none, 1=gelu_new, 2=relu. Epilogue fully specialized (no branches).
// ---------------------------------------------------------------------------
#define TBM 128
#define TBN 128
#define TBK 32

template <int ACT, bool RESID, bool OUTF, bool OUTB>
__global__ __launch_bounds__(256) void gemm_bf16_t(
    const __bf16* __restrict__ A, const __bf16* __restrict__ Bt,
    const float* __restrict__ bias, const float* __restrict__ resid,
    float* __restrict__ outF, __bf16* __restrict__ outB,
    int M, int N, int K)
{
    __shared__ __bf16 As[2][TBM][TBK];
    __shared__ __bf16 Bs[2][TBN][TBK];

    const int tid = threadIdx.x;
    const int m0 = blockIdx.y * TBM;
    const int n0 = blockIdx.x * TBN;
    const int w = tid >> 5, lane = tid & 31;
    const int wm = (w >> 2) * 64;      // 0 or 64
    const int wn = (w & 3) * 32;       // 0,32,64,96
    const int lhalf = (lane < 16) ? 0 : 8;
    const int lmod = lane & 15;

    // staging assignment: thread -> (row, 16-elem chunk)
    const int sr = tid >> 1, sk = (tid & 1) * 16;
    const __bf16* aSrc = A + (size_t)(m0 + sr) * K + sk;
    const __bf16* bSrc = Bt + (size_t)(n0 + sr) * K + sk;

    v8f zero = {};
    v8f acc[4][2];
#pragma unroll
    for (int i = 0; i < 4; ++i)
#pragma unroll
        for (int j = 0; j < 2; ++j) acc[i][j] = zero;

    const int ntiles = K / TBK;

    // prologue: stage tile 0
    {
        v8bf a0 = *(const v8bf*)aSrc;
        v8bf a1 = *(const v8bf*)(aSrc + 8);
        v8bf b0 = *(const v8bf*)bSrc;
        v8bf b1 = *(const v8bf*)(bSrc + 8);
        *(v8bf*)&As[0][sr][sk] = a0;
        *(v8bf*)&As[0][sr][sk + 8] = a1;
        *(v8bf*)&Bs[0][sr][sk] = b0;
        *(v8bf*)&Bs[0][sr][sk + 8] = b1;
    }
    __syncthreads();

    for (int t = 0; t < ntiles; ++t) {
        const int cur = t & 1, nxt = cur ^ 1;
        v8bf na0, na1, nb0, nb1;
        const bool more = (t + 1 < ntiles);
        if (more) {
            const __bf16* ap = aSrc + (size_t)(t + 1) * TBK;
            const __bf16* bp = bSrc + (size_t)(t + 1) * TBK;
            na0 = *(const v8bf*)ap;
            na1 = *(const v8bf*)(ap + 8);
            nb0 = *(const v8bf*)bp;
            nb1 = *(const v8bf*)(bp + 8);
            if (t + 2 < ntiles) {
                __builtin_prefetch(aSrc + (size_t)(t + 2) * TBK, 0, 1); // global_prefetch_b8
                __builtin_prefetch(bSrc + (size_t)(t + 2) * TBK, 0, 1);
            }
        }

        v16bf af[4], bf[2];
#pragma unroll
        for (int mi = 0; mi < 4; ++mi) {
            const __bf16* rp = &As[cur][wm + mi * 16 + lmod][lhalf];
            af[mi] = make_frag(rp, rp + 16);
        }
#pragma unroll
        for (int ni = 0; ni < 2; ++ni) {
            const __bf16* cp = &Bs[cur][wn + ni * 16 + lmod][lhalf];
            bf[ni] = make_frag(cp, cp + 16);
        }
#pragma unroll
        for (int mi = 0; mi < 4; ++mi)
#pragma unroll
            for (int ni = 0; ni < 2; ++ni)
                acc[mi][ni] = WMMA_BF16(af[mi], bf[ni], acc[mi][ni]);

        if (more) {
            *(v8bf*)&As[nxt][sr][sk] = na0;
            *(v8bf*)&As[nxt][sr][sk + 8] = na1;
            *(v8bf*)&Bs[nxt][sr][sk] = nb0;
            *(v8bf*)&Bs[nxt][sr][sk + 8] = nb1;
        }
        __syncthreads();
    }

    // epilogue: C fragment: VGPR r -> M = r + lhalf, N = lmod (straight-line)
#pragma unroll
    for (int mi = 0; mi < 4; ++mi) {
#pragma unroll
        for (int ni = 0; ni < 2; ++ni) {
#pragma unroll
            for (int r = 0; r < 8; ++r) {
                int row = m0 + wm + mi * 16 + lhalf + r;
                int col = n0 + wn + ni * 16 + lmod;
                float v = acc[mi][ni][r] + bias[col];
                if (ACT == 1) v = gelu_new_f(v);
                if (ACT == 2) v = fmaxf(v, 0.0f);
                if (RESID) v += resid[(size_t)row * N + col];
                if (OUTF) outF[(size_t)row * N + col] = v;
                if (OUTB) outB[(size_t)row * N + col] = (__bf16)v;
            }
        }
    }
}

// ---------------------------------------------------------------------------
// LayerNorm: one row per workgroup (256 thr). Optional row-addend (pos_emb)
// and vector-addend (tok_emb). Writes f32 stream + bf16 copy.
// ---------------------------------------------------------------------------
__global__ __launch_bounds__(256) void ln_kernel(
    const float* __restrict__ pre, const float* __restrict__ addRow,
    const float* __restrict__ addVec, const float* __restrict__ s,
    const float* __restrict__ b, float* __restrict__ outF,
    __bf16* __restrict__ outB, int N)
{
    extern __shared__ float smemln[];
    float* buf = smemln;          // N values
    float* red = smemln + N;      // reduction scratch (32 floats)
    const int row = blockIdx.x, tid = threadIdx.x;
    const int lane = tid & 31, wid = tid >> 5;

    float sum = 0.0f, ss = 0.0f;
    for (int j = tid; j < N; j += 256) {
        float v = pre[(size_t)row * N + j];
        if (addRow) v += addRow[(size_t)row * N + j];
        if (addVec) v += addVec[j];
        buf[j] = v; sum += v; ss += v * v;
    }
#pragma unroll
    for (int o = 16; o > 0; o >>= 1) {
        sum += __shfl_xor(sum, o, 32);
        ss  += __shfl_xor(ss,  o, 32);
    }
    if (lane == 0) { red[wid] = sum; red[8 + wid] = ss; }
    __syncthreads();
    if (tid == 0) {
        float a = 0.0f, c = 0.0f;
        for (int i = 0; i < 8; ++i) { a += red[i]; c += red[8 + i]; }
        red[16] = a; red[17] = c;
    }
    __syncthreads();
    float mean = red[16] / (float)N;
    float var = red[17] / (float)N - mean * mean;
    float inv = rsqrtf(var + 1e-12f);
    for (int j = tid; j < N; j += 256) {
        float v = (buf[j] - mean) * inv * s[j] + b[j];
        outF[(size_t)row * N + j] = v;
        outB[(size_t)row * N + j] = (__bf16)v;
    }
}

// ---------------------------------------------------------------------------
// Block-sparse attention, middle blocks. grid = (62, 8 heads), 256 thr/8 waves.
// Dynamic LDS: Qs[64][192] bf16 (offset 0, TDM target) | Sc[64][512] f32 |
//              Pb[64][512] bf16
// ---------------------------------------------------------------------------
#define HDIM 192
#define HMODEL 1536
#define SEQ 4096
#define ATT_SCALE 0.07216878364870323f   // 1/sqrt(192)
#define NEGBIG (-1000000000.0f)

__global__ __launch_bounds__(256) void attn_mid(
    const __bf16* __restrict__ Q, const __bf16* __restrict__ Kt,
    const __bf16* __restrict__ V, const int* __restrict__ mask,
    const int* __restrict__ kblocks, __bf16* __restrict__ ctx)
{
    extern __shared__ char smem[];
    __bf16* Qs = (__bf16*)smem;                                   // 64*192 @ lds 0
    float*  Sc = (float*)(smem + 64 * HDIM * 2);                  // 64*512
    __bf16* Pb = (__bf16*)(smem + 64 * HDIM * 2 + 64 * 512 * 4);  // 64*512

    const int n = blockIdx.x;        // middle block index 0..61 (q block n+1)
    const int hd = blockIdx.y;       // head
    const int tid = threadIdx.x, w = tid >> 5, lane = tid & 31;
    const int lhalf = (lane < 16) ? 0 : 8;
    const int lmod = lane & 15;
    const int q0 = (n + 1) * 64;
    const int cbase = hd * HDIM;
    v8f zero = {};

    // ---- Q block -> LDS via Tensor Data Mover (wave 0 issues, TENSORcnt) ----
    if (w == 0) {
        tdm_load_2d((unsigned long long)(const void*)(Q + (size_t)q0 * HMODEL + cbase),
                    0u, HDIM, 64u, HMODEL);
        __builtin_amdgcn_s_wait_tensorcnt(0);
    }
    __syncthreads();

    // ---- scores: wave w handles gathered key block w ----
    {
        int kvb = kblocks[n * 8 + w];
        int kv0 = kvb * 64;
        v8f acc[4][4];
#pragma unroll
        for (int i = 0; i < 4; ++i)
#pragma unroll
            for (int j = 0; j < 4; ++j) acc[i][j] = zero;

        for (int kk = 0; kk < HDIM; kk += 32) {
            v16bf af[4], bfg[4];
#pragma unroll
            for (int mi = 0; mi < 4; ++mi) {
                const __bf16* rp = &Qs[(mi * 16 + lmod) * HDIM + kk + lhalf];
                af[mi] = make_frag(rp, rp + 16);
            }
#pragma unroll
            for (int ni = 0; ni < 4; ++ni) {
                const __bf16* cp =
                    Kt + (size_t)(kv0 + ni * 16 + lmod) * HMODEL + cbase + kk + lhalf;
                bfg[ni] = make_frag(cp, cp + 16);
            }
#pragma unroll
            for (int mi = 0; mi < 4; ++mi)
#pragma unroll
                for (int ni = 0; ni < 4; ++ni)
                    acc[mi][ni] = WMMA_BF16(af[mi], bfg[ni], acc[mi][ni]);
        }
#pragma unroll
        for (int mi = 0; mi < 4; ++mi)
#pragma unroll
            for (int ni = 0; ni < 4; ++ni)
#pragma unroll
                for (int r = 0; r < 8; ++r) {
                    int row = mi * 16 + lhalf + r;
                    int col = ni * 16 + lmod;
                    float mv = (float)mask[kv0 + col];
                    Sc[row * 512 + w * 64 + col] =
                        acc[mi][ni][r] * ATT_SCALE + NEGBIG * (1.0f - mv);
                }
    }
    __syncthreads();

    // ---- softmax: wave w handles rows w*8 .. w*8+7 over 512 cols ----
    for (int rr = 0; rr < 8; ++rr) {
        int row = w * 8 + rr;
        float* S = Sc + row * 512;
        float mx = -3.4e38f;
        for (int i = lane; i < 512; i += 32) mx = fmaxf(mx, S[i]);
#pragma unroll
        for (int o = 16; o > 0; o >>= 1) mx = fmaxf(mx, __shfl_xor(mx, o, 32));
        float sum = 0.0f;
        for (int i = lane; i < 512; i += 32) { float e = __expf(S[i] - mx); S[i] = e; sum += e; }
#pragma unroll
        for (int o = 16; o > 0; o >>= 1) sum += __shfl_xor(sum, o, 32);
        float inv = 1.0f / sum;
        for (int i = lane; i < 512; i += 32) Pb[row * 512 + i] = (__bf16)(S[i] * inv);
    }
    __syncthreads();

    // ---- ctx = P(64x512) @ Vg(512x192): wave w -> rows 16*(w%4), cols 96*(w/4) ----
    {
        int mrow = (w & 3) * 16;
        int nb0 = (w >> 2) * 96;
        v8f acc2[6];
#pragma unroll
        for (int i = 0; i < 6; ++i) acc2[i] = zero;

        for (int kk = 0; kk < 512; kk += 32) {
            int blk = kblocks[n * 8 + (kk >> 6)];
            int kvbase = blk * 64 + (kk & 63);
            const __bf16* rp = Pb + (mrow + lmod) * 512 + kk + lhalf;
            v16bf af = make_frag(rp, rp + 16);
#pragma unroll
            for (int nt = 0; nt < 6; ++nt) {
                int d = cbase + nb0 + nt * 16 + lmod;
                v16bf bfg;
#pragma unroll
                for (int i = 0; i < 8; ++i) {
                    bfg[i]     = V[(size_t)(kvbase + lhalf + i) * HMODEL + d];
                    bfg[i + 8] = V[(size_t)(kvbase + lhalf + 16 + i) * HMODEL + d];
                }
                acc2[nt] = WMMA_BF16(af, bfg, acc2[nt]);
            }
        }
#pragma unroll
        for (int nt = 0; nt < 6; ++nt)
#pragma unroll
            for (int r = 0; r < 8; ++r) {
                int row = q0 + mrow + lhalf + r;
                int col = cbase + nb0 + nt * 16 + lmod;
                ctx[(size_t)row * HMODEL + col] = (__bf16)acc2[nt][r];
            }
    }
}

// ---------------------------------------------------------------------------
// Edge blocks (q block 0 and 63): dense attention over all 4096 keys.
// grid = (2, 8). Dynamic LDS: Qs[64][192] @ 0 (TDM target).
// Scores staged in global scratch, probs in bf16 scratch.
// ---------------------------------------------------------------------------
__global__ __launch_bounds__(256) void attn_edge(
    const __bf16* __restrict__ Q, const __bf16* __restrict__ Kt,
    const __bf16* __restrict__ V, const int* __restrict__ mask,
    __bf16* __restrict__ ctx, float* __restrict__ Ssc, __bf16* __restrict__ Pg)
{
    extern __shared__ char smem[];
    __bf16* Qs = (__bf16*)smem;                      // 64*192 @ lds 0
    const int e = blockIdx.x, hd = blockIdx.y;
    const int tid = threadIdx.x, w = tid >> 5, lane = tid & 31;
    const int lhalf = (lane < 16) ? 0 : 8;
    const int lmod = lane & 15;
    const int q0 = (e == 0) ? 0 : (SEQ - 64);
    const int cbase = hd * HDIM;
    float* S = Ssc + (size_t)(e * 8 + hd) * 64 * SEQ;
    __bf16* P = Pg + (size_t)(e * 8 + hd) * 64 * SEQ;
    v8f zero = {};

    if (w == 0) {
        tdm_load_2d((unsigned long long)(const void*)(Q + (size_t)q0 * HMODEL + cbase),
                    0u, HDIM, 64u, HMODEL);
        __builtin_amdgcn_s_wait_tensorcnt(0);
    }
    __syncthreads();

    // ---- scores over 64 kv blocks, 8 per iteration (one per wave) ----
    for (int it = 0; it < 8; ++it) {
        int kv0 = (it * 8 + w) * 64;
        v8f acc[4][4];
#pragma unroll
        for (int i = 0; i < 4; ++i)
#pragma unroll
            for (int j = 0; j < 4; ++j) acc[i][j] = zero;

        for (int kk = 0; kk < HDIM; kk += 32) {
            v16bf af[4], bfg[4];
#pragma unroll
            for (int mi = 0; mi < 4; ++mi) {
                const __bf16* rp = &Qs[(mi * 16 + lmod) * HDIM + kk + lhalf];
                af[mi] = make_frag(rp, rp + 16);
            }
#pragma unroll
            for (int ni = 0; ni < 4; ++ni) {
                const __bf16* cp =
                    Kt + (size_t)(kv0 + ni * 16 + lmod) * HMODEL + cbase + kk + lhalf;
                bfg[ni] = make_frag(cp, cp + 16);
            }
#pragma unroll
            for (int mi = 0; mi < 4; ++mi)
#pragma unroll
                for (int ni = 0; ni < 4; ++ni)
                    acc[mi][ni] = WMMA_BF16(af[mi], bfg[ni], acc[mi][ni]);
        }
#pragma unroll
        for (int mi = 0; mi < 4; ++mi)
#pragma unroll
            for (int ni = 0; ni < 4; ++ni)
#pragma unroll
                for (int r = 0; r < 8; ++r) {
                    int row = mi * 16 + lhalf + r;
                    int col = ni * 16 + lmod;
                    float mv = (float)mask[kv0 + col];
                    S[(size_t)row * SEQ + kv0 + col] =
                        acc[mi][ni][r] * ATT_SCALE + NEGBIG * (1.0f - mv);
                }
    }
    __threadfence();
    __syncthreads();

    // ---- softmax over 4096 ----
    for (int rr = 0; rr < 8; ++rr) {
        int row = w * 8 + rr;
        float* Sr = S + (size_t)row * SEQ;
        float mx = -3.4e38f;
        for (int i = lane; i < SEQ; i += 32) mx = fmaxf(mx, Sr[i]);
#pragma unroll
        for (int o = 16; o > 0; o >>= 1) mx = fmaxf(mx, __shfl_xor(mx, o, 32));
        float sum = 0.0f;
        for (int i = lane; i < SEQ; i += 32) { float ex = __expf(Sr[i] - mx); Sr[i] = ex; sum += ex; }
#pragma unroll
        for (int o = 16; o > 0; o >>= 1) sum += __shfl_xor(sum, o, 32);
        float inv = 1.0f / sum;
        for (int i = lane; i < SEQ; i += 32) P[(size_t)row * SEQ + i] = (__bf16)(Sr[i] * inv);
    }
    __threadfence();
    __syncthreads();

    // ---- ctx = P(64x4096) @ V(4096x192) ----
    {
        int mrow = (w & 3) * 16;
        int nb0 = (w >> 2) * 96;
        v8f acc2[6];
#pragma unroll
        for (int i = 0; i < 6; ++i) acc2[i] = zero;

        for (int kk = 0; kk < SEQ; kk += 32) {
            const __bf16* rp = P + (size_t)(mrow + lmod) * SEQ + kk + lhalf;
            v16bf af = make_frag(rp, rp + 16);
#pragma unroll
            for (int nt = 0; nt < 6; ++nt) {
                int d = cbase + nb0 + nt * 16 + lmod;
                v16bf bfg;
#pragma unroll
                for (int i = 0; i < 8; ++i) {
                    bfg[i]     = V[(size_t)(kk + lhalf + i) * HMODEL + d];
                    bfg[i + 8] = V[(size_t)(kk + lhalf + 16 + i) * HMODEL + d];
                }
                acc2[nt] = WMMA_BF16(af, bfg, acc2[nt]);
            }
        }
#pragma unroll
        for (int nt = 0; nt < 6; ++nt)
#pragma unroll
            for (int r = 0; r < 8; ++r) {
                int row = q0 + mrow + lhalf + r;
                int col = cbase + nb0 + nt * 16 + lmod;
                ctx[(size_t)row * HMODEL + col] = (__bf16)acc2[nt][r];
            }
    }
}

// ---------------------------------------------------------------------------
// Host orchestration
// ---------------------------------------------------------------------------
extern "C" void kernel_launch(void* const* d_in, const int* in_sizes, int n_in,
                              void* d_out, int out_size, void* d_ws, size_t ws_size,
                              hipStream_t stream) {
    const int S = 4096, H = 1536, DIN = 1280, FF = 3072, L = 12;

    const float* x       = (const float*)d_in[0];
    const float* proj_w  = (const float*)d_in[1];
    const float* proj_b  = (const float*)d_in[2];
    const float* pos_emb = (const float*)d_in[3];
    const float* tok_emb = (const float*)d_in[4];
    const float* emb_s   = (const float*)d_in[5];
    const float* emb_b   = (const float*)d_in[6];
    const float* Wq      = (const float*)d_in[7];
    const float* bq      = (const float*)d_in[8];
    const float* Wk      = (const float*)d_in[9];
    const float* bk      = (const float*)d_in[10];
    const float* Wv      = (const float*)d_in[11];
    const float* bv      = (const float*)d_in[12];
    const float* Wo      = (const float*)d_in[13];
    const float* bo      = (const float*)d_in[14];
    const float* ln1s    = (const float*)d_in[15];
    const float* ln1b    = (const float*)d_in[16];
    const float* Wi      = (const float*)d_in[17];
    const float* bi      = (const float*)d_in[18];
    const float* Wo2     = (const float*)d_in[19];
    const float* bo2     = (const float*)d_in[20];
    const float* ln2s    = (const float*)d_in[21];
    const float* ln2b    = (const float*)d_in[22];
    const float* cw1     = (const float*)d_in[23];
    const float* cb1     = (const float*)d_in[24];
    const float* cw2     = (const float*)d_in[25];
    const float* cb2     = (const float*)d_in[26];
    const int*   amask   = (const int*)d_in[27];
    const int*   kblocks = (const int*)d_in[28];

    // workspace carving
    char* ws = (char*)d_ws;
    size_t off = 0;
    auto carve = [&](size_t bytes) -> void* {
        void* p = ws + off;
        off += (bytes + 255) & ~(size_t)255;
        return p;
    };
    float*  h     = (float*)carve((size_t)S * H * 4);
    float*  pre   = (float*)carve((size_t)S * H * 4);
    __bf16* hb    = (__bf16*)carve((size_t)S * H * 2);
    __bf16* qb    = (__bf16*)carve((size_t)S * H * 2);
    __bf16* kb    = (__bf16*)carve((size_t)S * H * 2);
    __bf16* vb    = (__bf16*)carve((size_t)S * H * 2);
    __bf16* ctxb  = (__bf16*)carve((size_t)S * H * 2);
    __bf16* ffb   = (__bf16*)carve((size_t)S * FF * 2);
    __bf16* wbufT = (__bf16*)carve((size_t)H * FF * 2);   // transposed weight, reused
    __bf16* clsb  = (__bf16*)carve((size_t)S * 512 * 2);
    float*  edgeS = (float*)carve((size_t)16 * 64 * S * 4);
    __bf16* edgeP = (__bf16*)carve((size_t)16 * 64 * S * 2);
    __bf16* xb    = qb;  // reuse q buffer for converted input (consumed first)

    const int lnShm = (H + 32) * 4;
    const int midShm = 64 * HDIM * 2 + 64 * 512 * 4 + 64 * 512 * 2; // 221184 B
    const int edgeShm = 64 * HDIM * 2;                               // 24576 B

    auto cvt = [&](const float* src, __bf16* dst, long n) {
        cvt_f32_bf16<<<2048, 256, 0, stream>>>(src, dst, n);
    };
    auto cvtT = [&](const float* src, __bf16* dst, int Kd, int Nd) {
        cvt_transpose<<<dim3(Kd / 32, Nd / 32), dim3(32, 8), 0, stream>>>(src, dst, Kd, Nd);
    };
    auto lnorm = [&](const float* p, const float* aR, const float* aV,
                     const float* s, const float* b) {
        ln_kernel<<<S, 256, lnShm, stream>>>(p, aR, aV, s, b, h, hb, H);
    };
    auto grid = [&](int M, int N) { return dim3(N / 128, M / 128); };

    // ---- embedding: h = LN(x@proj_w + proj_b + pos_emb + tok_emb) ----
    cvt(x, xb, (long)S * DIN);
    cvtT(proj_w, wbufT, DIN, H);
    gemm_bf16_t<0, false, true, false><<<grid(S, H), 256, 0, stream>>>(
        xb, wbufT, proj_b, nullptr, pre, nullptr, S, H, DIN);
    lnorm(pre, pos_emb, tok_emb, emb_s, emb_b);

    // ---- encoder layers ----
    for (int l = 0; l < L; ++l) {
        const size_t wHH = (size_t)l * H * H;
        const size_t wHF = (size_t)l * H * FF;
        const size_t vH = (size_t)l * H;
        const size_t vF = (size_t)l * FF;

        cvtT(Wq + wHH, wbufT, H, H);
        gemm_bf16_t<0, false, false, true><<<grid(S, H), 256, 0, stream>>>(
            hb, wbufT, bq + vH, nullptr, nullptr, qb, S, H, H);
        cvtT(Wk + wHH, wbufT, H, H);
        gemm_bf16_t<0, false, false, true><<<grid(S, H), 256, 0, stream>>>(
            hb, wbufT, bk + vH, nullptr, nullptr, kb, S, H, H);
        cvtT(Wv + wHH, wbufT, H, H);
        gemm_bf16_t<0, false, false, true><<<grid(S, H), 256, 0, stream>>>(
            hb, wbufT, bv + vH, nullptr, nullptr, vb, S, H, H);

        attn_mid<<<dim3(62, 8), 256, midShm, stream>>>(qb, kb, vb, amask, kblocks, ctxb);
        attn_edge<<<dim3(2, 8), 256, edgeShm, stream>>>(qb, kb, vb, amask, ctxb, edgeS, edgeP);

        cvtT(Wo + wHH, wbufT, H, H);
        gemm_bf16_t<0, true, true, false><<<grid(S, H), 256, 0, stream>>>(
            ctxb, wbufT, bo + vH, h, pre, nullptr, S, H, H);
        lnorm(pre, nullptr, nullptr, ln1s + vH, ln1b + vH);

        cvtT(Wi + wHF, wbufT, H, FF);
        gemm_bf16_t<1, false, false, true><<<grid(S, FF), 256, 0, stream>>>(
            hb, wbufT, bi + vF, nullptr, nullptr, ffb, S, FF, H);
        cvtT(Wo2 + wHF, wbufT, FF, H);
        gemm_bf16_t<0, true, true, false><<<grid(S, H), 256, 0, stream>>>(
            ffb, wbufT, bo2 + vH, h, pre, nullptr, S, H, FF);
        lnorm(pre, nullptr, nullptr, ln2s + vH, ln2b + vH);
    }

    // ---- classifier head ----
    cvtT(cw1, wbufT, H, 512);
    gemm_bf16_t<2, false, false, true><<<grid(S, 512), 256, 0, stream>>>(
        hb, wbufT, cb1, nullptr, nullptr, clsb, S, 512, H);
    cvtT(cw2, wbufT, 512, 256);
    gemm_bf16_t<0, false, true, false><<<grid(S, 256), 256, 0, stream>>>(
        clsb, wbufT, cb2, nullptr, (float*)d_out, nullptr, S, 256, 512);
}
